// GRULanguageModel_90795608637819
// MI455X (gfx1250) — compile-verified, matrix operand order checked by reading, not run on hardware
//
#include <hip/hip_runtime.h>
#include <hip/hip_bf16.h>

// Problem sizes (fixed by the reference).
#define B_   64
#define T_   512
#define E_   512
#define H_   1024
#define G3_  3072
#define V_   256
#define BT_  (B_ * T_)

typedef __attribute__((ext_vector_type(16))) __bf16 v16bf;
typedef __attribute__((ext_vector_type(8)))  float  v8f;
typedef unsigned int v4u  __attribute__((ext_vector_type(4)));
typedef int          v8i_t __attribute__((ext_vector_type(8)));
typedef int          v4i_t __attribute__((ext_vector_type(4)));

union Frag16 { unsigned u[8]; v16bf v; };

__device__ __forceinline__ unsigned short bf_bits(float f) {
    union { float f; unsigned u; } v; v.f = f;
    unsigned u = v.u;
    unsigned r = (u + 0x7FFFu + ((u >> 16) & 1u)) >> 16;  // round-to-nearest-even
    return (unsigned short)r;
}
__device__ __forceinline__ __bf16 f2bf(float f) {
    unsigned short s = bf_bits(f);
    __bf16 b; __builtin_memcpy(&b, &s, 2); return b;
}
__device__ __forceinline__ float bf2f(__bf16 b) {
    unsigned short s; __builtin_memcpy(&s, &b, 2);
    union { unsigned u; float f; } v; v.u = ((unsigned)s) << 16; return v.f;
}

// CDNA5 16-bit A/B fragment K index for VGPR v (ISA 7.12.2):
// lanes 0-15: V0..3 -> K pairs {0..7}, V4..7 -> {16..23}; lanes 16-31: +8.
__device__ __forceinline__ int frag_k(int v, int lane) {
    int k = (v < 4) ? (2 * v) : (16 + 2 * (v - 4));
    return k + ((lane >= 16) ? 8 : 0);
}

// --- CDNA5 async global->LDS (per-lane, ASYNCcnt) ---------------------------
__device__ __forceinline__ void async_b128(const __bf16* gsrc, __bf16* lds_dst) {
    unsigned loff = (unsigned)(size_t)lds_dst;   // flat->LDS truncation rule
    unsigned long long ga = (unsigned long long)(size_t)gsrc;
    asm volatile("global_load_async_to_lds_b128 %0, %1, off"
                 :: "v"(loff), "v"(ga) : "memory");
}
__device__ __forceinline__ void wait_async0() {
    asm volatile("s_wait_asynccnt 0" ::: "memory");
}

// --- CDNA5 Tensor Data Mover descriptors (CDNA5 ISA) --------
// Group 0: count=1 | lds_addr | global_addr[56:0] | type=2.
__device__ __forceinline__ v4u tdm_group0(unsigned lds_off, const void* gaddr) {
    unsigned long long ga = (unsigned long long)(size_t)gaddr;
    v4u g;
    g.x = 1u;                                        // count=1, user D#
    g.y = lds_off;                                   // lds_addr (bytes)
    g.z = (unsigned)ga;                              // global_addr[31:0]
    g.w = (unsigned)((ga >> 32) & 0x1FFFFFFull) | 0x80000000u; // [56:32]|type=2
    return g;
}
// Group 1: flags | tensor_dim0/1 | tile_dim0/1/2 | dim0/dim1 strides.
__device__ __forceinline__ v8i_t tdm_group1(unsigned flags,
                                            unsigned tdim0, unsigned tdim1,
                                            unsigned tile0, unsigned tile1,
                                            unsigned tile2,
                                            unsigned long long s0,
                                            unsigned long long s1) {
    v8i_t g;
    g[0] = (int)flags;                               // mask|dsize|pad cfg
    g[1] = (int)((tdim0 & 0xFFFFu) << 16);           // abar=0 | tdim0[15:0]
    g[2] = (int)((tdim0 >> 16) | ((tdim1 & 0xFFFFu) << 16));
    g[3] = (int)((tdim1 >> 16) | (tile0 << 16));
    g[4] = (int)(tile1 | (tile2 << 16));
    g[5] = (int)(unsigned)(s0 & 0xFFFFFFFFull);
    g[6] = (int)(unsigned)(((s0 >> 32) & 0xFFFFu) | ((s1 & 0xFFFFu) << 16));
    g[7] = (int)(unsigned)((s1 >> 16) & 0xFFFFFFFFull);
    return g;
}
// data_size=2B (code 1). Padded variant: pad after 16 DWORDs (code 3) by
// 4 DWORDs (code 3) -> produces the [n][40] bank-padded LDS rows natively.
#define TDM_FLAGS_2B        (1u << 16)
#define TDM_FLAGS_2B_PAD    ((1u << 16) | (1u << 20) | (3u << 22) | (3u << 25))

// ---------------------------------------------------------------------------
// bf16 GEMM:  C(MxN) = A(MxK) @ W(NxK)^T + bias     (W in native N-major)
// mode 0: store bf16 row-major C.   mode 1: store f32 with (t*B+b) -> (b,t)
// row remap into logits (B,T,V).   M%128==0, N%128==0, K%32==0.
// Tiles move via per-lane async global->LDS copies (ASYNCcnt).
// ---------------------------------------------------------------------------
__global__ __launch_bounds__(256)
void gemm_bf16_wmma(const __bf16* __restrict__ A, const __bf16* __restrict__ W,
                    const float* __restrict__ bias,
                    __bf16* __restrict__ Cbf, float* __restrict__ Cf,
                    int M, int N, int K, int mode)
{
    __shared__ __align__(16) __bf16 As[2][128][32];   // [m][k]
    __shared__ __align__(16) __bf16 Bs[2][128][40];   // [n][k], padded

    const int tid  = threadIdx.x;
    const int lane = tid & 31;
    const int wave = tid >> 5;
    const int m0   = blockIdx.y * 128;
    const int n0   = blockIdx.x * 128;
    const int wm   = wave * 16;
    const int mrow = lane & 15;

    v8f zero8 = {0.f,0.f,0.f,0.f,0.f,0.f,0.f,0.f};
    v8f acc[8];
#pragma unroll
    for (int i = 0; i < 8; i++) acc[i] = zero8;

    auto loadTiles = [&](int buf, int k0) {
#pragma unroll
        for (int i = tid; i < 512; i += 256) {
            int row = i >> 2;
            int seg = (i & 3) * 8;
            async_b128(A + (size_t)(m0 + row) * K + k0 + seg, &As[buf][row][seg]);
            async_b128(W + (size_t)(n0 + row) * K + k0 + seg, &Bs[buf][row][seg]);
        }
    };

    const int nsteps = K >> 5;
    loadTiles(0, 0);
    wait_async0();
    __syncthreads();

    for (int s = 0; s < nsteps; s++) {
        int buf = s & 1;
        if (s + 1 < nsteps) loadTiles(buf ^ 1, (s + 1) << 5);

        Frag16 fa, fb[8];
#pragma unroll
        for (int v = 0; v < 8; v++)
            fa.u[v] = *(const unsigned*)&As[buf][wm + mrow][frag_k(v, lane)];
#pragma unroll
        for (int nt = 0; nt < 8; nt++) {
            int ncol = nt * 16 + mrow;
#pragma unroll
            for (int v = 0; v < 8; v++)
                fb[nt].u[v] = *(const unsigned*)&Bs[buf][ncol][frag_k(v, lane)];
        }
#pragma unroll
        for (int nt = 0; nt < 8; nt++)
            acc[nt] = __builtin_amdgcn_wmma_f32_16x16x32_bf16(
                false, fa.v, false, fb[nt].v, (short)0, acc[nt], false, false);

        wait_async0();
        __syncthreads();
    }

    const int radd = (lane >> 4) << 3;
#pragma unroll
    for (int nt = 0; nt < 8; nt++) {
        int n = n0 + nt * 16 + mrow;
        float bv = bias ? bias[n] : 0.0f;
#pragma unroll
        for (int r = 0; r < 8; r++) {
            int m = m0 + wm + r + radd;
            float val = acc[nt][r] + bv;
            if (mode == 0) {
                Cbf[(size_t)m * N + n] = f2bf(val);
            } else {
                int b = m & (B_ - 1);   // m = t*B + b
                int t = m >> 6;
                Cf[((size_t)b * T_ + t) * V_ + n] = val;
            }
        }
    }
}

// ---------------------------------------------------------------------------
// One GRU timestep.  Grid: H/32 WGs; each owns 32 hidden columns and computes
// gh = h_prev @ Whh^T for all 3 gates (M=64, K=1024, N=32x3) with WMMA, then
// the GRU update in f32.  Tiles are fetched by the Tensor Data Mover:
//   A: 2D tile 32(K) x 64(rows, stride H)            -> As (unpadded)
//   B: 3D tile 32(K) x 32(rows, stride H) x 3(gates, stride H*H)
//      with TDM pad-on-load producing the [n][40] padded rows -> Bs
// Wave 0 issues both descriptors, waits on TENSORcnt, barrier publishes LDS.
// ---------------------------------------------------------------------------
__global__ __launch_bounds__(256)
void gru_step(const float*  __restrict__ hprev_f, const __bf16* __restrict__ hprev_b,
              const __bf16* __restrict__ Whhb,     // (3H x H) row-major bf16
              const __bf16* __restrict__ Gi,       // (T*B) x 3H, bih already added
              const float*  __restrict__ bhh,      // 3H
              float* __restrict__ hnext_f, __bf16* __restrict__ hnext_b,
              __bf16* __restrict__ Hall,           // (T*B) x H output trace
              int t)
{
    __shared__ __align__(16) __bf16 As[2][64][32];       // h_prev tile [m][k]
    __shared__ __align__(16) __bf16 Bs[2][3][32][40];    // [gate][n][k] padded

    const int tid  = threadIdx.x;
    const int lane = tid & 31;
    const int wave = tid >> 5;
    const int j0   = blockIdx.x * 32;
    const int mt   = wave & 3;
    const int nt   = wave >> 2;
    const int mrow = lane & 15;

    v8f zero8 = {0.f,0.f,0.f,0.f,0.f,0.f,0.f,0.f};
    v8f acc[3];
#pragma unroll
    for (int g = 0; g < 3; g++) acc[g] = zero8;

    const v4i_t z4 = {0, 0, 0, 0};
    const v8i_t z8 = {0, 0, 0, 0, 0, 0, 0, 0};

    auto loadTiles = [&](int buf, int kk) {
        if (wave == 0) {
            // A: h_prev rows 0..63, cols kk..kk+31.
            v4u a0 = tdm_group0((unsigned)(size_t)&As[buf][0][0], hprev_b + kk);
            v8i_t a1 = tdm_group1(TDM_FLAGS_2B, H_, B_, 32, 64, 0,
                                  (unsigned long long)H_, 0ull);
            __builtin_amdgcn_tensor_load_to_lds(a0, a1, z4, z4, z8, 0);
            // B: Whh rows {g*H + j0 + n}, cols kk..kk+31, 3D with pad-on-load.
            v4u b0 = tdm_group0((unsigned)(size_t)&Bs[buf][0][0][0],
                                Whhb + (size_t)j0 * H_ + kk);
            v8i_t b1 = tdm_group1(TDM_FLAGS_2B_PAD, H_, G3_, 32, 32, 3,
                                  (unsigned long long)H_,
                                  (unsigned long long)H_ * H_);
            v4i_t b2 = {3, 0, 0, 0};                 // tensor_dim2 = 3 gates
            __builtin_amdgcn_tensor_load_to_lds(b0, b1, b2, z4, z8, 0);
        }
    };
    auto tensorWait = [&]() {
        if (wave == 0) __builtin_amdgcn_s_wait_tensorcnt(0);
    };

    loadTiles(0, 0);
    tensorWait();
    __syncthreads();

    for (int s = 0; s < (H_ >> 5); s++) {
        int buf = s & 1;
        if (s + 1 < (H_ >> 5)) loadTiles(buf ^ 1, (s + 1) << 5);

        Frag16 fa, fb[3];
#pragma unroll
        for (int v = 0; v < 8; v++)
            fa.u[v] = *(const unsigned*)&As[buf][mt * 16 + mrow][frag_k(v, lane)];
#pragma unroll
        for (int g = 0; g < 3; g++)
#pragma unroll
            for (int v = 0; v < 8; v++)
                fb[g].u[v] =
                    *(const unsigned*)&Bs[buf][g][nt * 16 + mrow][frag_k(v, lane)];
#pragma unroll
        for (int g = 0; g < 3; g++)
            acc[g] = __builtin_amdgcn_wmma_f32_16x16x32_bf16(
                false, fa.v, false, fb[g].v, (short)0, acc[g], false, false);

        tensorWait();
        __syncthreads();
    }

    // GRU update for this lane's 8 rows x 1 column.
    const int j    = j0 + nt * 16 + mrow;
    const int radd = (lane >> 4) << 3;
    const float bhr = bhh[j], bhz = bhh[H_ + j], bhn = bhh[2 * H_ + j];
#pragma unroll
    for (int r = 0; r < 8; r++) {
        int b = mt * 16 + r + radd;                       // batch row
        size_t gbase = ((size_t)t * B_ + b) * G3_;
        float gir = bf2f(Gi[gbase + j]);
        float giz = bf2f(Gi[gbase + H_ + j]);
        float gin = bf2f(Gi[gbase + 2 * H_ + j]);
        float ghr = acc[0][r] + bhr;
        float ghz = acc[1][r] + bhz;
        float ghn = acc[2][r] + bhn;
        float rr = 1.0f / (1.0f + __expf(-(gir + ghr)));
        float zz = 1.0f / (1.0f + __expf(-(giz + ghz)));
        float nn = tanhf(gin + rr * ghn);
        float hp = hprev_f[(size_t)b * H_ + j];
        float hn = (1.0f - zz) * nn + zz * hp;
        hnext_f[(size_t)b * H_ + j] = hn;
        __bf16 hb = f2bf(hn);
        hnext_b[(size_t)b * H_ + j] = hb;
        Hall[((size_t)t * B_ + b) * H_ + j] = hb;
    }
}

// ---------------------------------------------------------------------------
// Helpers: weight cast (native layout), embedding gather, state zeroing.
// ---------------------------------------------------------------------------
__global__ void cast_bf16(const float* __restrict__ W, __bf16* __restrict__ Wb, int n)
{
    int i = blockIdx.x * blockDim.x + threadIdx.x;
    if (i < n) Wb[i] = f2bf(W[i]);
}

__global__ void embed_gather(const int* __restrict__ x, const float* __restrict__ embed,
                             __bf16* __restrict__ Xemb)
{   // Xemb[(t*B+b)*E + e] = embed[x[b*T+t]][e]  (time-major for the scan)
    int i = blockIdx.x * blockDim.x + threadIdx.x;
    if (i >= BT_ * E_) return;
    int e = i % E_;
    int m = i / E_;            // m = t*B + b
    int b = m & (B_ - 1);
    int t = m >> 6;
    int xi = x[b * T_ + t];
    Xemb[i] = f2bf(embed[(size_t)xi * E_ + e]);
}

__global__ void zero_state(float* f0, float* f1, __bf16* b0, __bf16* b1)
{
    int i = blockIdx.x * blockDim.x + threadIdx.x;
    if (i < B_ * H_) {
        f0[i] = 0.0f; f1[i] = 0.0f;
        __bf16 z = f2bf(0.0f);
        b0[i] = z; b1[i] = z;
    }
}

// ---------------------------------------------------------------------------
// Host orchestration.
// ---------------------------------------------------------------------------
extern "C" void kernel_launch(void* const* d_in, const int* in_sizes, int n_in,
                              void* d_out, int out_size, void* d_ws, size_t ws_size,
                              hipStream_t stream)
{
    const int*   x     = (const int*)  d_in[0];
    const float* embed = (const float*)d_in[1];
    const float* Wih0  = (const float*)d_in[2];
    const float* Whh0  = (const float*)d_in[3];
    const float* bih0  = (const float*)d_in[4];
    const float* bhh0  = (const float*)d_in[5];
    const float* Wih1  = (const float*)d_in[6];
    const float* Whh1  = (const float*)d_in[7];
    const float* bih1  = (const float*)d_in[8];
    const float* bhh1  = (const float*)d_in[9];
    const float* W_out = (const float*)d_in[10];
    const float* b_out = (const float*)d_in[11];
    float* out = (float*)d_out;

    // Bump allocator over d_ws (total ~400 MB).
    char* ws = (char*)d_ws;
    size_t off = 0;
    auto alloc = [&](size_t bytes) -> void* {
        off = (off + 255) & ~(size_t)255;
        void* p = ws + off;
        off += bytes;
        return p;
    };
    __bf16* Wih0b = (__bf16*)alloc((size_t)G3_ * E_ * 2);
    __bf16* Whh0b = (__bf16*)alloc((size_t)G3_ * H_ * 2);
    __bf16* Wih1b = (__bf16*)alloc((size_t)G3_ * H_ * 2);
    __bf16* Whh1b = (__bf16*)alloc((size_t)G3_ * H_ * 2);
    __bf16* Woutb = (__bf16*)alloc((size_t)V_  * H_ * 2);
    __bf16* Xemb  = (__bf16*)alloc((size_t)BT_ * E_ * 2);
    __bf16* Gi    = (__bf16*)alloc((size_t)BT_ * G3_ * 2);   // reused L0 then L1
    __bf16* H0all = (__bf16*)alloc((size_t)BT_ * H_ * 2);
    __bf16* H1all = (__bf16*)alloc((size_t)BT_ * H_ * 2);
    float*  h0f   = (float*) alloc((size_t)2 * B_ * H_ * 4);
    float*  h1f   = (float*) alloc((size_t)2 * B_ * H_ * 4);
    __bf16* h0b   = (__bf16*)alloc((size_t)2 * B_ * H_ * 2);
    __bf16* h1b   = (__bf16*)alloc((size_t)2 * B_ * H_ * 2);
    const size_t BH = (size_t)B_ * H_;

    // 1) Weight precast to bf16 in native (N x K) layout.
    cast_bf16<<<(G3_ * E_ + 255) / 256, 256, 0, stream>>>(Wih0, Wih0b, G3_ * E_);
    cast_bf16<<<(G3_ * H_ + 255) / 256, 256, 0, stream>>>(Whh0, Whh0b, G3_ * H_);
    cast_bf16<<<(G3_ * H_ + 255) / 256, 256, 0, stream>>>(Wih1, Wih1b, G3_ * H_);
    cast_bf16<<<(G3_ * H_ + 255) / 256, 256, 0, stream>>>(Whh1, Whh1b, G3_ * H_);
    cast_bf16<<<(V_  * H_ + 255) / 256, 256, 0, stream>>>(W_out, Woutb, V_ * H_);

    // 2) Embedding gather (time-major) + zero initial hidden state.
    embed_gather<<<(BT_ * E_ + 255) / 256, 256, 0, stream>>>(x, embed, Xemb);
    zero_state<<<(B_ * H_ + 255) / 256, 256, 0, stream>>>(h0f, h1f, h0b, h1b);

    // 3) Layer 0 input GEMM for all timesteps: Gi = Xemb @ Wih0^T + bih0.
    dim3 gBig(G3_ / 128, BT_ / 128);
    gemm_bf16_wmma<<<gBig, 256, 0, stream>>>(Xemb, Wih0b, bih0, Gi, nullptr,
                                             BT_, G3_, E_, 0);

    // 4) Layer 0 recurrence (ping-pong hidden state).
    for (int t = 0; t < T_; t++) {
        int p = t & 1, q = (t + 1) & 1;
        gru_step<<<H_ / 32, 256, 0, stream>>>(
            h0f + p * BH, h0b + p * BH, Whh0b, Gi, bhh0,
            h0f + q * BH, h0b + q * BH, H0all, t);
    }

    // 5) Layer 1 input GEMM: Gi = H0all @ Wih1^T + bih1 (reuse Gi buffer).
    gemm_bf16_wmma<<<gBig, 256, 0, stream>>>(H0all, Wih1b, bih1, Gi, nullptr,
                                             BT_, G3_, H_, 0);

    // 6) Layer 1 recurrence.
    for (int t = 0; t < T_; t++) {
        int p = t & 1, q = (t + 1) & 1;
        gru_step<<<H_ / 32, 256, 0, stream>>>(
            h1f + p * BH, h1b + p * BH, Whh1b, Gi, bhh1,
            h1f + q * BH, h1b + q * BH, H1all, t);
    }

    // 7) Output projection with fused (t,b)->(b,t) remap: logits f32.
    dim3 gOut(V_ / 128, BT_ / 128);
    gemm_bf16_wmma<<<gOut, 256, 0, stream>>>(H1all, Woutb, b_out, nullptr, out,
                                             BT_, V_, H_, 1);
    (void)in_sizes; (void)n_in; (void)out_size; (void)ws_size;
}
